// Qwen3MoeSparseMoeBlock_3109556322625
// MI455X (gfx1250) — matrix operand order repaired and checked by reading, not compiled, not run on hardware
//
#include <hip/hip_runtime.h>
#include <hip/hip_bf16.h>

// ---- problem constants (Qwen3-MoE config from reference) ----
constexpr int kD   = 2048;   // hidden
constexpr int kE   = 64;     // experts
constexpr int kI   = 768;    // moe intermediate
constexpr int kTOP = 8;      // top-k
constexpr int kM   = 4096;   // tokens (B*S)
constexpr int kCAP = 1024;   // per-expert capacity

typedef unsigned short u16;
typedef unsigned int   u32;
typedef __attribute__((ext_vector_type(16))) __bf16 v16bf;
typedef __attribute__((ext_vector_type(8)))  float  v8f;
typedef __attribute__((ext_vector_type(4)))  float  f4;
typedef __attribute__((ext_vector_type(4)))  u16    us4;
typedef __attribute__((ext_vector_type(4)))  int    v4i;   // async-LDS builtin ptr type

union FragBF { v16bf v; f4 q[2]; };

#if __has_builtin(__builtin_amdgcn_global_load_async_to_lds_b128) && \
    __has_builtin(__builtin_amdgcn_s_wait_asynccnt)
#define USE_ASYNC_LDS 1
#endif

__device__ __forceinline__ u16 f2bf(float f) {            // f32 -> bf16 RNE
  unsigned u = __float_as_uint(f);
  u += 0x7FFFu + ((u >> 16) & 1u);
  return (u16)(u >> 16);
}
__device__ __forceinline__ u32 pack2bf(float lo, float hi) {
  return (u32)f2bf(lo) | ((u32)f2bf(hi) << 16);
}

// ---------------- router: logits -> softmax -> top8 -> renorm ----------------
__global__ __launch_bounds__(64)
void moe_router(const float* __restrict__ x, const float* __restrict__ rw,
                int* __restrict__ topi, float* __restrict__ topw) {
  const int t = blockIdx.x, e = threadIdx.x;
  const float* xr = x + (size_t)t * kD;
  float acc = 0.f;
  for (int d = 0; d < kD; ++d) acc = fmaf(xr[d], rw[(size_t)d * kE + e], acc);
  __shared__ float sl[kE];
  __shared__ float sp[kE];
  __shared__ float smax;
  sl[e] = acc;
  __syncthreads();
  if (e == 0) { float mx = sl[0]; for (int i = 1; i < kE; ++i) mx = fmaxf(mx, sl[i]); smax = mx; }
  __syncthreads();
  sp[e] = __expf(sl[e] - smax);
  __syncthreads();
  if (e == 0) {
    float denom = 0.f;
    for (int i = 0; i < kE; ++i) denom += sp[i];
    float pv[kTOP]; int pi[kTOP]; float wsum = 0.f;
    for (int k = 0; k < kTOP; ++k) {
      float best = -1.f; int bi = 0;
      for (int i = 0; i < kE; ++i) { float v = sp[i]; if (v > best) { best = v; bi = i; } }
      sp[bi] = -2.f;
      pv[k] = best / denom; pi[k] = bi; wsum += pv[k];
    }
    const float inv = 1.f / wsum;
    for (int k = 0; k < kTOP; ++k) { topi[t * kTOP + k] = pi[k]; topw[t * kTOP + k] = pv[k] * inv; }
  }
}

__global__ void moe_zero_counts(int* counts) { if (threadIdx.x < kE) counts[threadIdx.x] = 0; }

// ---------------- slot assignment (atomic per-expert append) ----------------
__global__ void moe_assign(const int* __restrict__ topi, int* __restrict__ counts,
                           int* __restrict__ slotarr, int* __restrict__ tokmap) {
  int a = blockIdx.x * blockDim.x + threadIdx.x;
  if (a >= kM * kTOP) return;
  int e = topi[a];
  int slot = atomicAdd(&counts[e], 1);
  if (slot < kCAP) { slotarr[a] = slot; tokmap[e * kCAP + slot] = a; }
  else slotarr[a] = -1;
}

// ---------------- dispatch: gather token rows, f32 -> bf16 ----------------
__global__ __launch_bounds__(64)
void moe_dispatch(const float* __restrict__ x, const int* __restrict__ counts,
                  const int* __restrict__ tokmap, u16* __restrict__ xe) {
  const int b = blockIdx.x;
  const int e = b >> 10, slot = b & (kCAP - 1);
  int cnt = counts[e]; cnt = cnt < kCAP ? cnt : kCAP;
  if (slot >= cnt) return;
  const int a = tokmap[(size_t)e * kCAP + slot];
  const int t = a >> 3;
  const float* src = x + (size_t)t * kD;
  u16* dst = xe + ((size_t)e * kCAP + slot) * kD;
  #pragma unroll
  for (int i = 0; i < 8; ++i) {
    int d = (i * 64 + (int)threadIdx.x) * 4;
    f4 v = *(const f4*)(src + d);
    us4 o = { f2bf(v[0]), f2bf(v[1]), f2bf(v[2]), f2bf(v[3]) };
    *(us4*)(dst + d) = o;
  }
}

// ---------------- fused gate/up GEMM + SiLU*up, bf16 WMMA ----------------
// block tile: 128(M) x 64(N), KT=32. 8 waves: (wm 0..3) x (wn 0..1), each 32x32
// per matrix. A staged to LDS (async if available); weights transposed to LDS
// as bf16 with paired-dword stores.
#define LSTR 40   // halfword stride: keeps ds_load_b128 16B-aligned, de-conflicts banks

__global__ __launch_bounds__(256)
void moe_gateup(const u16* __restrict__ xe, const float* __restrict__ gatew,
                const float* __restrict__ upw, const int* __restrict__ counts,
                u16* __restrict__ hbuf) {
  const int e = blockIdx.x >> 3;
  const int mtile = blockIdx.x & 7;
  int cnt = counts[e]; cnt = cnt < kCAP ? cnt : kCAP;
  const int rowbase = mtile * 128;
  if (rowbase >= cnt) return;               // uniform early-out: skip empty tiles
  const int nbase = blockIdx.y * 64;

  __shared__ __align__(16) u16 lA[128 * LSTR];
  __shared__ __align__(16) u16 lG[64 * LSTR];
  __shared__ __align__(16) u16 lU[64 * LSTR];

  const int tid = threadIdx.x;
  const int lane = tid & 31, wave = tid >> 5;
  const int wm = wave & 3, wn = wave >> 2;
  const int r = lane & 15, hi = lane >> 4;

  v8f cg[2][2] = {}, cu[2][2] = {};

  const u16*  ab = xe + ((size_t)e * kCAP + rowbase) * kD;
  const float* gb = gatew + (size_t)e * kD * kI + nbase;
  const float* ub = upw   + (size_t)e * kD * kI + nbase;

  // staging decomposition: A as 512 x 16B chunks (2/thread);
  // weights as 2(k) x 4(n) sub-blocks per thread -> paired b32 LDS stores.
  const int k0 = (tid >> 4) * 2, n0 = (tid & 15) * 4;

  for (int kb = 0; kb < kD; kb += 32) {
    #pragma unroll
    for (int c = 0; c < 2; ++c) {
      int idx = tid + c * 256;
      int row = idx >> 2, kk = (idx & 3) * 8;
#ifdef USE_ASYNC_LDS
      __builtin_amdgcn_global_load_async_to_lds_b128(
          (v4i*)(ab + (size_t)row * kD + kb + kk), (v4i*)&lA[row * LSTR + kk], 0, 0);
#else
      *(f4*)&lA[row * LSTR + kk] = *(const f4*)(ab + (size_t)row * kD + kb + kk);
#endif
    }
    {
      const float* gp = gb + (size_t)(kb + k0) * kI + n0;
      const float* up = ub + (size_t)(kb + k0) * kI + n0;
      f4 g0 = *(const f4*)gp, g1 = *(const f4*)(gp + kI);
      f4 u0 = *(const f4*)up, u1 = *(const f4*)(up + kI);
      #pragma unroll
      for (int j = 0; j < 4; ++j) {
        *(u32*)&lG[(n0 + j) * LSTR + k0] = pack2bf(g0[j], g1[j]);
        *(u32*)&lU[(n0 + j) * LSTR + k0] = pack2bf(u0[j], u1[j]);
      }
    }
#ifdef USE_ASYNC_LDS
    __builtin_amdgcn_s_wait_asynccnt(0);
#endif
    __syncthreads();
    // build fragments in contiguous k order (k = hi*16 + s), same for A and B
    FragBF a[2], bg[2], bu[2];
    #pragma unroll
    for (int mi = 0; mi < 2; ++mi) {
      const f4* p = (const f4*)&lA[(wm * 32 + mi * 16 + r) * LSTR + hi * 16];
      a[mi].q[0] = p[0];  // K = hi*16 .. +7
      a[mi].q[1] = p[1];  // K = hi*16+8 .. +15
    }
    #pragma unroll
    for (int ni = 0; ni < 2; ++ni) {
      const f4* pg = (const f4*)&lG[(wn * 32 + ni * 16 + r) * LSTR + hi * 16];
      bg[ni].q[0] = pg[0]; bg[ni].q[1] = pg[1];  // K = hi*16 .. +15
      const f4* pu = (const f4*)&lU[(wn * 32 + ni * 16 + r) * LSTR + hi * 16];
      bu[ni].q[0] = pu[0]; bu[ni].q[1] = pu[1];
    }
    #pragma unroll
    for (int mi = 0; mi < 2; ++mi)
      #pragma unroll
      for (int ni = 0; ni < 2; ++ni) {
        cg[mi][ni] = __builtin_amdgcn_wmma_f32_16x16x32_bf16(
            false, a[mi].v, false, bg[ni].v, (short)0, cg[mi][ni], false, false);
        cu[mi][ni] = __builtin_amdgcn_wmma_f32_16x16x32_bf16(
            false, a[mi].v, false, bu[ni].v, (short)0, cu[mi][ni], false, false);
      }
    __syncthreads();
  }
  // epilogue: h = silu(gate) * up -> bf16  (v_rcp_f32-based sigmoid, no IEEE div)
  #pragma unroll
  for (int mi = 0; mi < 2; ++mi)
    #pragma unroll
    for (int ni = 0; ni < 2; ++ni)
      #pragma unroll
      for (int j = 0; j < 8; ++j) {
        int m = wm * 32 + mi * 16 + hi * 8 + j;
        int n = wn * 32 + ni * 16 + r;
        float g = cg[mi][ni][j], u = cu[mi][ni][j];
        float sig = __builtin_amdgcn_rcpf(1.f + __expf(-g));
        float hv = g * sig * u;
        hbuf[((size_t)e * kCAP + rowbase + m) * kI + nbase + n] = f2bf(hv);
      }
}

// ---------------- down GEMM: ye = h @ down_proj (K = 768) ----------------
__global__ __launch_bounds__(256)
void moe_down(const u16* __restrict__ hbuf, const float* __restrict__ downw,
              const int* __restrict__ counts, float* __restrict__ ye) {
  const int e = blockIdx.x >> 3;
  const int mtile = blockIdx.x & 7;
  int cnt = counts[e]; cnt = cnt < kCAP ? cnt : kCAP;
  const int rowbase = mtile * 128;
  if (rowbase >= cnt) return;
  const int nbase = blockIdx.y * 64;

  __shared__ __align__(16) u16 lA[128 * LSTR];
  __shared__ __align__(16) u16 lB[64 * LSTR];

  const int tid = threadIdx.x;
  const int lane = tid & 31, wave = tid >> 5;
  const int wm = wave & 3, wn = wave >> 2;
  const int r = lane & 15, hi = lane >> 4;

  v8f cc[2][2] = {};
  const u16*  ab = hbuf + ((size_t)e * kCAP + rowbase) * kI;
  const float* bb = downw + (size_t)e * kI * kD + nbase;
  const int k0 = (tid >> 4) * 2, n0 = (tid & 15) * 4;

  for (int kb = 0; kb < kI; kb += 32) {
    #pragma unroll
    for (int c = 0; c < 2; ++c) {
      int idx = tid + c * 256;
      int row = idx >> 2, kk = (idx & 3) * 8;
#ifdef USE_ASYNC_LDS
      __builtin_amdgcn_global_load_async_to_lds_b128(
          (v4i*)(ab + (size_t)row * kI + kb + kk), (v4i*)&lA[row * LSTR + kk], 0, 0);
#else
      *(f4*)&lA[row * LSTR + kk] = *(const f4*)(ab + (size_t)row * kI + kb + kk);
#endif
    }
    {
      const float* bp = bb + (size_t)(kb + k0) * kD + n0;
      f4 b0 = *(const f4*)bp, b1 = *(const f4*)(bp + kD);
      #pragma unroll
      for (int j = 0; j < 4; ++j)
        *(u32*)&lB[(n0 + j) * LSTR + k0] = pack2bf(b0[j], b1[j]);
    }
#ifdef USE_ASYNC_LDS
    __builtin_amdgcn_s_wait_asynccnt(0);
#endif
    __syncthreads();
    FragBF a[2], b[2];
    #pragma unroll
    for (int mi = 0; mi < 2; ++mi) {
      const f4* p = (const f4*)&lA[(wm * 32 + mi * 16 + r) * LSTR + hi * 16];
      a[mi].q[0] = p[0]; a[mi].q[1] = p[1];
    }
    #pragma unroll
    for (int ni = 0; ni < 2; ++ni) {
      const f4* pb = (const f4*)&lB[(wn * 32 + ni * 16 + r) * LSTR + hi * 16];
      b[ni].q[0] = pb[0]; b[ni].q[1] = pb[1];
    }
    #pragma unroll
    for (int mi = 0; mi < 2; ++mi)
      #pragma unroll
      for (int ni = 0; ni < 2; ++ni)
        cc[mi][ni] = __builtin_amdgcn_wmma_f32_16x16x32_bf16(
            false, a[mi].v, false, b[ni].v, (short)0, cc[mi][ni], false, false);
    __syncthreads();
  }
  #pragma unroll
  for (int mi = 0; mi < 2; ++mi)
    #pragma unroll
    for (int ni = 0; ni < 2; ++ni)
      #pragma unroll
      for (int j = 0; j < 8; ++j) {
        int m = wm * 32 + mi * 16 + hi * 8 + j;
        int n = wn * 32 + ni * 16 + r;
        ye[((size_t)e * kCAP + rowbase + m) * kD + nbase + n] = cc[mi][ni][j];
      }
}

// ---------------- combine: fixed-order weighted sum over top-8 ----------------
__global__ __launch_bounds__(256)
void moe_combine(const int* __restrict__ topi, const float* __restrict__ topw,
                 const int* __restrict__ slotarr, const float* __restrict__ ye,
                 float* __restrict__ out) {
  const int t = blockIdx.x, tid = threadIdx.x;
  const float* src[kTOP]; float wk[kTOP];
  #pragma unroll
  for (int k = 0; k < kTOP; ++k) {
    int a = t * kTOP + k;
    int s = slotarr[a];
    int e = topi[a];
    wk[k] = (s >= 0) ? topw[a] : 0.f;
    int sc = (s >= 0) ? s : 0;
    src[k] = ye + ((size_t)e * kCAP + sc) * kD;
  }
  for (int d = tid; d < kD; d += 256) {
    float acc = 0.f;
    #pragma unroll
    for (int k = 0; k < kTOP; ++k) acc = fmaf(wk[k], src[k][d], acc);
    out[(size_t)t * kD + d] = acc;
  }
}

// ---------------- host launcher ----------------
extern "C" void kernel_launch(void* const* d_in, const int* in_sizes, int n_in,
                              void* d_out, int out_size, void* d_ws, size_t ws_size,
                              hipStream_t stream) {
  (void)in_sizes; (void)n_in; (void)out_size; (void)ws_size;
  const float* x     = (const float*)d_in[0];
  const float* rw    = (const float*)d_in[1];
  const float* gatew = (const float*)d_in[2];
  const float* upw   = (const float*)d_in[3];
  const float* downw = (const float*)d_in[4];
  float* out = (float*)d_out;
  char*  ws  = (char*)d_ws;

  auto au = [](size_t v) { return (v + 255) & ~(size_t)255; };
  size_t off = 0;
  int*   counts  = (int*)(ws + off);  off += au((size_t)kE * 4);
  int*   topi    = (int*)(ws + off);  off += au((size_t)kM * kTOP * 4);
  float* topw    = (float*)(ws + off);off += au((size_t)kM * kTOP * 4);
  int*   slotarr = (int*)(ws + off);  off += au((size_t)kM * kTOP * 4);
  int*   tokmap  = (int*)(ws + off);  off += au((size_t)kE * kCAP * 4);
  u16*   xe      = (u16*)(ws + off);  off += au((size_t)kE * kCAP * kD * 2);
  u16*   hbuf    = (u16*)(ws + off);  off += au((size_t)kE * kCAP * kI * 2);
  float* ye      = (float*)(ws + off);

  moe_zero_counts<<<1, 64, 0, stream>>>(counts);
  moe_router<<<kM, 64, 0, stream>>>(x, rw, topi, topw);
  moe_assign<<<(kM * kTOP + 255) / 256, 256, 0, stream>>>(topi, counts, slotarr, tokmap);
  moe_dispatch<<<kE * kCAP, 64, 0, stream>>>(x, counts, tokmap, xe);
  moe_gateup<<<dim3(kE * 8, kI / 64), 256, 0, stream>>>(xe, gatew, upw, counts, hbuf);
  moe_down<<<dim3(kE * 8, kD / 64), 256, 0, stream>>>(hbuf, downw, counts, ye);
  moe_combine<<<kM, 256, 0, stream>>>(topi, topw, slotarr, ye, out);
}
